// glow_gru_coupling_layer_53824530153824
// MI455X (gfx1250) — matrix operand order repaired and checked by reading, not compiled, not run on hardware
//
#include <hip/hip_runtime.h>
#include <cstdint>
#include <cstddef>

// Problem constants (from reference): B=256, T=1024, C=128, D=64
#define B_SZ 256
#define T_SZ 1024
#define C_SZ 128
#define D_SZ 64
#define N3   192   // 3*D gate width
#define N2   128   // 2*D output width

typedef __attribute__((ext_vector_type(16))) __bf16       v16bf;
typedef __attribute__((ext_vector_type(8)))  float        v8f;
typedef __attribute__((ext_vector_type(4)))  unsigned int v4u;

union BF16Frag { v4u q[2]; v16bf v; };

// ---------------- LDS layout (bytes) ----------------
#define OFF_W    0         // 65536 bf16 weights = 131072 B
#define OFF_H1   131072    // 16x64 f32
#define OFF_H2   135168    // 16x64 f32
#define OFF_AX   139264    // 16x64 bf16 (A operand staging)
#define OFF_AH   141312    // 16x64 bf16 (A operand staging)
#define OFF_G    143360    // 16x192 f32 gates
#define OFF_GHN  155648    // 16x64 f32 (h@Wh n-gate part)
#define OFF_R    159744    // 16x128 f32 (Wo output)
#define OFF_X1   167936    // 16x64 f32
#define OFF_X2   172032    // 16x64 f32
#define OFF_BIAS 176128    // 1024 f32 biases
#define SMEM_BYTES 180224

// bf16 weight element offsets inside workspace / LDS (u16 units)
#define W_S2WX 0
#define W_S2WH 12288
#define W_S2WO 24576
#define W_S1WX 32768
#define W_S1WH 45056
#define W_S1WO 57344
#define W_TOTAL 65536

__device__ __forceinline__ unsigned short f2bf(float f) {
  unsigned int u = __float_as_uint(f);
  u += 0x7FFFu + ((u >> 16) & 1u);   // round-to-nearest-even
  return (unsigned short)(u >> 16);
}

__device__ __forceinline__ float sigmoid_f(float x) {
  return 1.0f / (1.0f + __expf(-x));
}

// e(s) = exp(CLAMP*0.636*atan(s/CLAMP)), CLAMP=5 -> 3.18*atan(0.2*s)
__device__ __forceinline__ float couple_e(float s) {
  return __expf(3.18f * atanf(0.2f * s));
}

// A fragment: 16x32 bf16 tile, A stored row-major [16][64] in LDS.
// Lane layout (ISA 7.12.2, 16-bit A 16x32): m = lane&15,
// k = kt*32 + (lane<16 ? 0 : 8) + {0..7}, then +16 for the upper 4 VGPRs.
__device__ __forceinline__ v16bf load_a_frag(const unsigned short* A, int lane, int kt) {
  int m  = lane & 15;
  int k0 = kt * 32 + ((lane >> 4) << 3);   // 0 or 8
  const unsigned short* p = A + m * 64 + k0;
  BF16Frag f;
  f.q[0] = *(const v4u*)(p);
  f.q[1] = *(const v4u*)(p + 16);
  return f.v;
}

// B fragment: 32x16 bf16 tile, weights stored COLUMN-major [N][64] bf16 in LDS.
// Lane layout (ISA 7.12.4 B pattern): n = lane&15,
// k = kt*32 + (lane<16 ? 0 : 8) + {0..7}, then +16 for the upper 4 VGPRs.
__device__ __forceinline__ v16bf load_b_frag(const unsigned short* Wc, int lane, int ntile, int kt) {
  int n  = ntile * 16 + (lane & 15);
  int k0 = kt * 32 + ((lane >> 4) << 3);   // 0 or 8
  const unsigned short* p = Wc + n * 64 + k0;
  BF16Frag f;
  f.q[0] = *(const v4u*)(p);
  f.q[1] = *(const v4u*)(p + 16);
  return f.v;
}

// Gate GEMM for one GRU stage: g[16][192] = aX@Wx + bx  (+ aH@Wh + bh fused
// for r/z columns); n-gate h-part kept separate in ghn[16][64].
__device__ __forceinline__ void gates_gemm(
    const unsigned short* sWx, const unsigned short* sWh,
    const unsigned short* aX, const unsigned short* aH,
    const float* bx, const float* bh,
    float* g, float* ghn, int wave, int lane) {
  const int cl   = lane & 15;
  const int mrow = (lane >> 4) << 3;   // 0 or 8

  // fused r/z tiles: wave and wave+4 (cols 0..127)
#pragma unroll
  for (int ti = 0; ti < 2; ++ti) {
    int ntile = wave + ti * 4;
    int col   = ntile * 16 + cl;
    float bias = bx[col] + bh[col];
    v8f acc;
#pragma unroll
    for (int i = 0; i < 8; ++i) acc[i] = bias;
#pragma unroll
    for (int kt = 0; kt < 2; ++kt) {
      v16bf a = load_a_frag(aX, lane, kt);
      v16bf b = load_b_frag(sWx, lane, ntile, kt);
      acc = __builtin_amdgcn_wmma_f32_16x16x32_bf16(false, a, false, b, (short)0, acc, false, false);
    }
#pragma unroll
    for (int kt = 0; kt < 2; ++kt) {
      v16bf a = load_a_frag(aH, lane, kt);
      v16bf b = load_b_frag(sWh, lane, ntile, kt);
      acc = __builtin_amdgcn_wmma_f32_16x16x32_bf16(false, a, false, b, (short)0, acc, false, false);
    }
#pragma unroll
    for (int r = 0; r < 8; ++r) g[(r + mrow) * N3 + col] = acc[r];
  }

  // n-gate tile: wave+8 (cols 128..191), keep x/h parts separate
  {
    int ntile = wave + 8;
    int col   = ntile * 16 + cl;
    int jcol  = col - 128;
    float biasx = bx[col];
    float biash = bh[col];
    v8f accX, accH;
#pragma unroll
    for (int i = 0; i < 8; ++i) { accX[i] = biasx; accH[i] = biash; }
#pragma unroll
    for (int kt = 0; kt < 2; ++kt) {
      v16bf a = load_a_frag(aX, lane, kt);
      v16bf b = load_b_frag(sWx, lane, ntile, kt);
      accX = __builtin_amdgcn_wmma_f32_16x16x32_bf16(false, a, false, b, (short)0, accX, false, false);
    }
#pragma unroll
    for (int kt = 0; kt < 2; ++kt) {
      v16bf a = load_a_frag(aH, lane, kt);
      v16bf b = load_b_frag(sWh, lane, ntile, kt);
      accH = __builtin_amdgcn_wmma_f32_16x16x32_bf16(false, a, false, b, (short)0, accH, false, false);
    }
#pragma unroll
    for (int r = 0; r < 8; ++r) {
      g[(r + mrow) * N3 + col]   = accX[r];
      ghn[(r + mrow) * 64 + jcol] = accH[r];
    }
  }
}

// Output projection: r[16][128] = aH@Wo + bo
__device__ __forceinline__ void out_gemm(
    const unsigned short* sWo, const unsigned short* aH,
    const float* bo, float* rbuf, int wave, int lane) {
  const int cl   = lane & 15;
  const int mrow = (lane >> 4) << 3;
#pragma unroll
  for (int ti = 0; ti < 2; ++ti) {
    int ntile = wave + ti * 4;
    int col   = ntile * 16 + cl;
    float bias = bo[col];
    v8f acc;
#pragma unroll
    for (int i = 0; i < 8; ++i) acc[i] = bias;
#pragma unroll
    for (int kt = 0; kt < 2; ++kt) {
      v16bf a = load_a_frag(aH, lane, kt);
      v16bf b = load_b_frag(sWo, lane, ntile, kt);
      acc = __builtin_amdgcn_wmma_f32_16x16x32_bf16(false, a, false, b, (short)0, acc, false, false);
    }
#pragma unroll
    for (int r = 0; r < 8; ++r) rbuf[(r + mrow) * N2 + col] = acc[r];
  }
}

// GRU elementwise update: reads fused gates g, n-gate h part ghn; updates
// h in place and writes bf16(h_next) into aOut.
__device__ __forceinline__ void gate_update(
    const float* g, const float* ghn, float* h, unsigned short* aOut, int tid) {
  for (int idx = tid; idx < 16 * 64; idx += 128) {
    int m = idx >> 6, j = idx & 63;
    float rg = sigmoid_f(g[m * N3 + j]);
    float zg = sigmoid_f(g[m * N3 + 64 + j]);
    float ng = tanhf(g[m * N3 + 128 + j] + rg * ghn[m * 64 + j]);
    float hn = (1.0f - zg) * ng + zg * h[idx];
    h[idx]   = hn;
    aOut[idx] = f2bf(hn);
  }
}

// fp32 row-major (K x N) -> bf16 column-major (N x 64); K is always 64.
__global__ void convert_w_kernel(const float* __restrict__ src, int N,
                                 unsigned short* __restrict__ dst) {
  int tid = blockIdx.x * blockDim.x + threadIdx.x;
  if (tid >= 64 * N) return;
  int n = tid >> 6, k = tid & 63;
  dst[tid] = f2bf(src[k * N + n]);
}

__global__ __launch_bounds__(128, 1)
void glow_gru_rnn_kernel(
    const float* __restrict__ x,
    const unsigned short* __restrict__ wbf,
    const float* __restrict__ s2bx, const float* __restrict__ s2bh, const float* __restrict__ s2bo,
    const float* __restrict__ s1bx, const float* __restrict__ s1bh, const float* __restrict__ s1bo,
    float* __restrict__ out) {
  extern __shared__ unsigned char smem[];
  unsigned short* sW  = (unsigned short*)(smem + OFF_W);
  float* h1   = (float*)(smem + OFF_H1);
  float* h2   = (float*)(smem + OFF_H2);
  unsigned short* aX = (unsigned short*)(smem + OFF_AX);
  unsigned short* aH = (unsigned short*)(smem + OFF_AH);
  float* g    = (float*)(smem + OFF_G);
  float* ghn  = (float*)(smem + OFF_GHN);
  float* rbuf = (float*)(smem + OFF_R);
  float* x1s  = (float*)(smem + OFF_X1);
  float* x2s  = (float*)(smem + OFF_X2);
  float* sb   = (float*)(smem + OFF_BIAS);

  const int tid  = threadIdx.x;
  const int wave = tid >> 5;
  const int lane = tid & 31;
  const int row0 = blockIdx.x * 16;

  // ---- block prologue: weights ws->LDS (b128 copies), biases, zero state ----
  {
    const v4u* src = (const v4u*)wbf;
    v4u* dst = (v4u*)sW;
    for (int i = tid; i < (W_TOTAL * 2) / 16; i += 128) dst[i] = src[i];
    for (int i = tid; i < N3; i += 128) {
      sb[i]       = s2bx[i];
      sb[192 + i] = s2bh[i];
      sb[512 + i] = s1bx[i];
      sb[704 + i] = s1bh[i];
    }
    if (tid < 128) { sb[384 + tid] = s2bo[tid]; sb[896 + tid] = s1bo[tid]; }
    for (int i = tid; i < 16 * 64; i += 128) { h1[i] = 0.0f; h2[i] = 0.0f; }
  }
  __syncthreads();

  const float* sb2x = sb + 0,   *sb2h = sb + 192, *sb2o = sb + 384;
  const float* sb1x = sb + 512, *sb1h = sb + 704, *sb1o = sb + 896;

  for (int t = 0; t < T_SZ; ++t) {
    // 1) load x tile; stage x2 as bf16 A operand, h2 as bf16 A operand
    for (int idx = tid; idx < 16 * C_SZ; idx += 128) {
      int m = idx >> 7, c = idx & 127;
      float v = x[((size_t)(row0 + m) * T_SZ + t) * C_SZ + c];
      if (c < 64) { x1s[m * 64 + c] = v; }
      else        { x2s[m * 64 + c - 64] = v; aX[m * 64 + c - 64] = f2bf(v); }
    }
    for (int idx = tid; idx < 16 * 64; idx += 128) aH[idx] = f2bf(h2[idx]);
    if (t + 1 < T_SZ && tid < 64) {   // prefetch next timestep (global_prefetch_b8)
      int m = tid >> 2, c = (tid & 3) << 5;
      __builtin_prefetch(&x[((size_t)(row0 + m) * T_SZ + (t + 1)) * C_SZ + c], 0, 1);
    }
    __syncthreads();

    // 2) stage-2 gate GEMMs: g = x2@Wx2 + h2@Wh2 (+biases), ghn = n-gate h part
    gates_gemm(sW + W_S2WX, sW + W_S2WH, aX, aH, sb2x, sb2h, g, ghn, wave, lane);
    __syncthreads();

    // 3) stage-2 GRU update: h2 <- h2n (in place), aX <- bf16(h2n)
    gate_update(g, ghn, h2, aX, tid);
    __syncthreads();

    // 4) r2 = h2n @ Wo2 + bo2
    out_gemm(sW + W_S2WO, aX, sb2o, rbuf, wave, lane);
    __syncthreads();

    // 5) coupling 1: y1 = e(s2v)*x1 + t2 -> out + aX; aH <- bf16(h1)
    for (int idx = tid; idx < 16 * 64; idx += 128) {
      int m = idx >> 6, j = idx & 63;
      float sv = rbuf[m * N2 + j];
      float tv = rbuf[m * N2 + 64 + j];
      float y1 = couple_e(sv) * x1s[idx] + tv;
      out[((size_t)(row0 + m) * T_SZ + t) * C_SZ + j] = y1;
      aX[idx] = f2bf(y1);
      aH[idx] = f2bf(h1[idx]);
    }
    __syncthreads();

    // 6) stage-1 gate GEMMs with A = y1, h1
    gates_gemm(sW + W_S1WX, sW + W_S1WH, aX, aH, sb1x, sb1h, g, ghn, wave, lane);
    __syncthreads();

    // 7) stage-1 GRU update: h1 <- h1n, aX <- bf16(h1n)
    gate_update(g, ghn, h1, aX, tid);
    __syncthreads();

    // 8) r1 = h1n @ Wo1 + bo1
    out_gemm(sW + W_S1WO, aX, sb1o, rbuf, wave, lane);
    __syncthreads();

    // 9) coupling 2: y2 = e(s1v)*x2 + t1 -> out
    for (int idx = tid; idx < 16 * 64; idx += 128) {
      int m = idx >> 6, j = idx & 63;
      float sv = rbuf[m * N2 + j];
      float tv = rbuf[m * N2 + 64 + j];
      float y2 = couple_e(sv) * x2s[idx] + tv;
      out[((size_t)(row0 + m) * T_SZ + t) * C_SZ + 64 + j] = y2;
    }
    __syncthreads();   // protect x2s/rbuf/aX before next iteration overwrites
  }
}

extern "C" void kernel_launch(void* const* d_in, const int* in_sizes, int n_in,
                              void* d_out, int out_size, void* d_ws, size_t ws_size,
                              hipStream_t stream) {
  (void)in_sizes; (void)n_in; (void)out_size; (void)ws_size;
  const float* x     = (const float*)d_in[0];
  const float* s1Wx  = (const float*)d_in[1];
  const float* s1Wh  = (const float*)d_in[2];
  const float* s1bx  = (const float*)d_in[3];
  const float* s1bh  = (const float*)d_in[4];
  const float* s1Wo  = (const float*)d_in[5];
  const float* s1bo  = (const float*)d_in[6];
  const float* s2Wx  = (const float*)d_in[7];
  const float* s2Wh  = (const float*)d_in[8];
  const float* s2bx  = (const float*)d_in[9];
  const float* s2bh  = (const float*)d_in[10];
  const float* s2Wo  = (const float*)d_in[11];
  const float* s2bo  = (const float*)d_in[12];
  float* out = (float*)d_out;
  unsigned short* wbf = (unsigned short*)d_ws;

  // convert fp32 row-major weights -> bf16 column-major in workspace
  convert_w_kernel<<<dim3(48), dim3(256), 0, stream>>>(s2Wx, N3, wbf + W_S2WX);
  convert_w_kernel<<<dim3(48), dim3(256), 0, stream>>>(s2Wh, N3, wbf + W_S2WH);
  convert_w_kernel<<<dim3(32), dim3(256), 0, stream>>>(s2Wo, N2, wbf + W_S2WO);
  convert_w_kernel<<<dim3(48), dim3(256), 0, stream>>>(s1Wx, N3, wbf + W_S1WX);
  convert_w_kernel<<<dim3(48), dim3(256), 0, stream>>>(s1Wh, N3, wbf + W_S1WH);
  convert_w_kernel<<<dim3(32), dim3(256), 0, stream>>>(s1Wo, N2, wbf + W_S1WO);

  static bool attr_set = []() {
    hipFuncSetAttribute((const void*)glow_gru_rnn_kernel,
                        hipFuncAttributeMaxDynamicSharedMemorySize, SMEM_BYTES);
    return true;
  }();
  (void)attr_set;

  glow_gru_rnn_kernel<<<dim3(B_SZ / 16), dim3(128), SMEM_BYTES, stream>>>(
      x, wbf, s2bx, s2bh, s2bo, s1bx, s1bh, s1bo, out);
}
